// GNAEEncoder_35605278883998
// MI455X (gfx1250) — matrix operand order repaired and checked by reading, not compiled, not run on hardware
//
#include <hip/hip_runtime.h>
#include <math.h>

typedef __attribute__((ext_vector_type(16))) __bf16 v16bf;
typedef __attribute__((ext_vector_type(8)))  float  v8f;

#define SCALE_F 1.8f
#define EPS_F   1e-12f

__device__ __forceinline__ void atomAddF(float* addr, float v) {
  // relaxed agent-scope fp32 add -> native global_atomic_add_f32 (no CAS loop)
  __hip_atomic_fetch_add(addr, v, __ATOMIC_RELAXED, __HIP_MEMORY_SCOPE_AGENT);
}

// ---- prep: convert W (64x256 f32) to bf16, init degree to 1.0 (self-loop) ----
__global__ void prep_kernel(const float* __restrict__ W, __bf16* __restrict__ Wb,
                            float* __restrict__ deg, int N) {
  int i = blockIdx.x * 256 + threadIdx.x;
  if (i < 64 * 256) Wb[i] = (__bf16)W[i];
  if (i < N) deg[i] = 1.0f;
}

// ---- in-degree via f32 atomics ----
__global__ void degree_kernel(const long long* __restrict__ dstI,
                              float* __restrict__ deg, int E) {
  int e = blockIdx.x * 256 + threadIdx.x;
  if (e < E) atomAddF(&deg[(int)dstI[e]], 1.0f);
}

// ---- deg -> deg^-1/2 (deg >= 1 always due to self-loops) ----
__global__ void dinv_kernel(float* __restrict__ deg, int N) {
  int i = blockIdx.x * 256 + threadIdx.x;
  if (i < N) deg[i] = 1.0f / sqrtf(deg[i]);
}

// ---- WMMA GEMM + L2-normalize + fold dinv[src]; writes g and out=g (self-loop) ----
__global__ void __launch_bounds__(128)
gemm_norm_kernel(const float* __restrict__ x, const __bf16* __restrict__ Wb,
                 const float* __restrict__ bias, const float* __restrict__ dinv,
                 float* __restrict__ g, float* __restrict__ out,
                 int N, int nTiles) {
  const int lane = threadIdx.x & 31;
  const int tile = blockIdx.x * 4 + (threadIdx.x >> 5);
  if (tile >= nTiles) return;              // wave-uniform: EXEC stays all-ones
  const int rowBase = tile * 16;
  const int half = lane >> 4;              // 0: lanes 0-15, 1: lanes 16-31
  const int lmod = lane & 15;

  int arow = rowBase + lmod;
  if (arow >= N) arow = N - 1;             // clamp (N=50000 is a multiple of 16 anyway)
  const float* xrow = x + (size_t)arow * 256;
  const __bf16* wbase = Wb + lmod * 256 + half * 16;

  v8f acc0 = {}, acc1 = {}, acc2 = {}, acc3 = {};

#pragma unroll
  for (int kb = 0; kb < 256; kb += 32) {
    // A fragment (16x32 bf16): lane holds K in [kb+half*16,+16) (contiguous)
    const float4* p0 = (const float4*)(xrow + kb + half * 16);
    const float4* p1 = (const float4*)(xrow + kb + half * 16 + 8);
    float4 a0 = p0[0], a1 = p0[1];
    float4 a2 = p1[0], a3 = p1[1];
    v16bf A;
    A[0]=(__bf16)a0.x;  A[1]=(__bf16)a0.y;  A[2]=(__bf16)a0.z;  A[3]=(__bf16)a0.w;
    A[4]=(__bf16)a1.x;  A[5]=(__bf16)a1.y;  A[6]=(__bf16)a1.z;  A[7]=(__bf16)a1.w;
    A[8]=(__bf16)a2.x;  A[9]=(__bf16)a2.y;  A[10]=(__bf16)a2.z; A[11]=(__bf16)a2.w;
    A[12]=(__bf16)a3.x; A[13]=(__bf16)a3.y; A[14]=(__bf16)a3.z; A[15]=(__bf16)a3.w;

    // B fragment (32x16 bf16): col = lmod, K = kb + half*16 + v (contiguous 32B)
    v16bf B0 = *(const v16bf*)(wbase + kb);
    v16bf B1 = *(const v16bf*)(wbase + kb + 16 * 256);
    v16bf B2 = *(const v16bf*)(wbase + kb + 32 * 256);
    v16bf B3 = *(const v16bf*)(wbase + kb + 48 * 256);

    acc0 = __builtin_amdgcn_wmma_f32_16x16x32_bf16(false, A, false, B0, (short)0, acc0, false, false);
    acc1 = __builtin_amdgcn_wmma_f32_16x16x32_bf16(false, A, false, B1, (short)0, acc1, false, false);
    acc2 = __builtin_amdgcn_wmma_f32_16x16x32_bf16(false, A, false, B2, (short)0, acc2, false, false);
    acc3 = __builtin_amdgcn_wmma_f32_16x16x32_bf16(false, A, false, B3, (short)0, acc3, false, false);
  }

  // epilogue: bias, row L2-norm (butterfly over 16 lanes per half), fold dinv
  float bn0 = bias[lmod], bn1 = bias[lmod + 16], bn2 = bias[lmod + 32], bn3 = bias[lmod + 48];
  float s[8];
#pragma unroll
  for (int r = 0; r < 8; ++r) {
    float h0 = acc0[r] + bn0, h1 = acc1[r] + bn1, h2 = acc2[r] + bn2, h3 = acc3[r] + bn3;
    acc0[r] = h0; acc1[r] = h1; acc2[r] = h2; acc3[r] = h3;
    s[r] = h0 * h0 + h1 * h1 + h2 * h2 + h3 * h3;
  }
#pragma unroll
  for (int r = 0; r < 8; ++r) {
    s[r] += __shfl_xor(s[r], 1, 32);
    s[r] += __shfl_xor(s[r], 2, 32);
    s[r] += __shfl_xor(s[r], 4, 32);
    s[r] += __shfl_xor(s[r], 8, 32);
  }
#pragma unroll
  for (int r = 0; r < 8; ++r) {
    int row = rowBase + half * 8 + r;      // D layout: m = r + half*8
    if (row >= N) continue;
    float sc = SCALE_F / fmaxf(sqrtf(s[r]), EPS_F) * dinv[row];
    int base = row * 64 + lmod;            // n = lmod + 16*ntile
    float v0 = acc0[r] * sc, v1 = acc1[r] * sc, v2 = acc2[r] * sc, v3 = acc3[r] * sc;
    g[base]      = v0; g[base + 16]   = v1; g[base + 32]   = v2; g[base + 48]   = v3;
    out[base]    = v0; out[base + 16] = v1; out[base + 32] = v2; out[base + 48] = v3;
  }
}

// ---- per-(edge,feature) scatter-add: out[dst] += g[src]; L2-resident atomics ----
__global__ void scatter_kernel(const long long* __restrict__ srcI,
                               const long long* __restrict__ dstI,
                               const float* __restrict__ g,
                               float* __restrict__ out, int E) {
  long long tid = (long long)blockIdx.x * blockDim.x + threadIdx.x;
  int e = (int)(tid >> 6);
  if (e >= E) return;
  int j = (int)(tid & 63);
  int s = (int)srcI[e];
  int d = (int)dstI[e];
  atomAddF(out + (size_t)d * 64 + j, g[(size_t)s * 64 + j]);
}

// ---- out[d,:] *= dinv[d] ----
__global__ void finalize_kernel(float* __restrict__ out, const float* __restrict__ dinv,
                                int total) {
  int i = blockIdx.x * 256 + threadIdx.x;
  if (i < total) out[i] *= dinv[i >> 6];
}

extern "C" void kernel_launch(void* const* d_in, const int* in_sizes, int n_in,
                              void* d_out, int out_size, void* d_ws, size_t ws_size,
                              hipStream_t stream) {
  const float* x        = (const float*)d_in[0];       // [N,256]
  const float* W        = (const float*)d_in[1];       // [64,256]
  const float* bias     = (const float*)d_in[2];       // [64]
  const long long* ei   = (const long long*)d_in[3];   // [2,E] int64
  int N = in_sizes[0] / 256;
  int E = in_sizes[3] / 2;
  const long long* srcI = ei;
  const long long* dstI = ei + E;
  float* out = (float*)d_out;

  // workspace layout: g [N*64] f32 | deg/dinv [N] f32 | Wbf16 [64*256]
  float* g    = (float*)d_ws;
  float* deg  = g + (size_t)N * 64;
  __bf16* Wb  = (__bf16*)(deg + N);

  int maxI = (N > 64 * 256) ? N : 64 * 256;
  prep_kernel<<<(maxI + 255) / 256, 256, 0, stream>>>(W, Wb, deg, N);
  degree_kernel<<<(E + 255) / 256, 256, 0, stream>>>(dstI, deg, E);
  dinv_kernel<<<(N + 255) / 256, 256, 0, stream>>>(deg, N);

  int nTiles = (N + 15) / 16;
  gemm_norm_kernel<<<(nTiles + 3) / 4, 128, 0, stream>>>(x, Wb, bias, deg, g, out, N, nTiles);

  long long totScatter = (long long)E * 64;
  scatter_kernel<<<(int)((totScatter + 255) / 256), 256, 0, stream>>>(srcI, dstI, g, out, E);
  finalize_kernel<<<(N * 64 + 255) / 256, 256, 0, stream>>>(out, deg, N * 64);
}
